// AttentionWithVillinaRelativePositionEncoding_10230612099275
// MI455X (gfx1250) — compile-verified, matrix operand order checked by reading, not run on hardware
//
#include <hip/hip_runtime.h>
#include <hip/hip_bf16.h>

// ---------------------------------------------------------------------------
// Relative-position attention, MI455X (gfx1250), wave32 + WMMA bf16 path.
//   1) f32->bf16 conversions (inputs, transposed weights, padded rel tables)
//   2) WMMA GEMM (K=64 per LDS stage, async global->LDS): Q/K/V projections
//   3) WMMA GEMM: R[bhq, j] = q . rel_key_table[j]  (rel-key bias table)
//   4) flash-attention kernel: scores WMMA + R-gather bias + online softmax
//      + P@V WMMA + LDS bucket W for rel-value + final W@RVt WMMA
//   5) WMMA GEMM: out = O @ Wo + bo  (f32 output)
// ---------------------------------------------------------------------------

typedef __bf16 bf16_t;
typedef __attribute__((ext_vector_type(16))) __bf16 v16bf;
typedef __attribute__((ext_vector_type(8)))  float  v8f;
typedef __attribute__((vector_size(16)))     int    v4i_t;

#define B_SZ   2
#define S_LEN  2048
#define H_DIM  1024
#define NH     16
#define HD     64
#define NTOK   (B_SZ * S_LEN)   // 4096
#define MAXREL 128
#define JTOT   257               // 2*MAXREL+1
#define JPAD   288               // 9*32 (WMMA K multiple)
#define RLD    320               // 5*64 (padded N for R GEMM / leading dim)

__device__ __forceinline__ bf16_t f2bf(float x) {
  unsigned u = __builtin_bit_cast(unsigned, x);
  unsigned r = u + 0x7FFFu + ((u >> 16) & 1u);      // round-to-nearest-even
  unsigned short h = (unsigned short)(r >> 16);
  return __builtin_bit_cast(bf16_t, h);
}

__device__ __forceinline__ v8f wmma_bf16(v16bf a, v16bf b, v8f c) {
  // D(f32 16x16) = A(bf16 16x32) * B(bf16 32x16) + C
  return __builtin_amdgcn_wmma_f32_16x16x32_bf16(false, a, false, b,
                                                 (short)0, c, false, false);
}

// 16-bit A-matrix fragment (16x32): lane half lh selects the K sub-block.
// Also used for B columns by reading rows of a transposed tile.
__device__ __forceinline__ v16bf frag_row16(const bf16_t* p, int lh) {
  v16bf r;
  const bf16_t* a = p + 8 * lh;
  const bf16_t* b = p + 16 + 8 * lh;
#pragma unroll
  for (int e = 0; e < 8; ++e) { r[e] = a[e]; r[e + 8] = b[e]; }
  return r;
}

// ---- CDNA5 async global->LDS copy (ASYNCcnt) with safe fallback -----------
#if __has_builtin(__builtin_amdgcn_global_load_async_to_lds_b128)
#define USE_ASYNC_LDS 1
#else
#define USE_ASYNC_LDS 0
#endif

__device__ __forceinline__ void g2l_b128(const void* g, void* l) {
#if USE_ASYNC_LDS
  // Builtin signature (per hipcc diagnostic): (v4i __device__*, v4i __shared__*, Ii, Ii)
  __builtin_amdgcn_global_load_async_to_lds_b128(
      (__attribute__((address_space(1))) v4i_t*)(g),
      (__attribute__((address_space(3))) v4i_t*)(l), 0, 0);
#else
  *(uint4*)l = *(const uint4*)g;
#endif
}

__device__ __forceinline__ void async_wait0() {
#if USE_ASYNC_LDS
#if __has_builtin(__builtin_amdgcn_s_wait_asynccnt)
  __builtin_amdgcn_s_wait_asynccnt(0);
#else
  asm volatile("s_wait_asynccnt 0" ::: "memory");
#endif
#endif
}

// ---------------------------- conversion kernels ---------------------------

__global__ void f32_to_bf16_kernel(const float* __restrict__ in,
                                   bf16_t* __restrict__ out, size_t n) {
  size_t i = (size_t)blockIdx.x * blockDim.x + threadIdx.x;
  if (i < n) out[i] = f2bf(in[i]);
}

__global__ void transpose_f32_to_bf16_kernel(const float* __restrict__ in,
                                             bf16_t* __restrict__ out,
                                             int rows, int cols) {
  size_t i = (size_t)blockIdx.x * blockDim.x + threadIdx.x;
  if (i < (size_t)rows * cols) {
    int r = (int)(i / cols), c = (int)(i % cols);
    out[(size_t)c * rows + r] = f2bf(in[i]);
  }
}

__global__ void pad_rows_bf16_kernel(const float* __restrict__ in,
                                     bf16_t* __restrict__ out,
                                     int inRows, int cols, int outRows) {
  size_t i = (size_t)blockIdx.x * blockDim.x + threadIdx.x;
  if (i < (size_t)outRows * cols) {
    int r = (int)(i / cols);
    out[i] = (r < inRows) ? f2bf(in[i]) : f2bf(0.f);
  }
}

__global__ void transpose_pad_bf16_kernel(const float* __restrict__ in,
                                          bf16_t* __restrict__ out,
                                          int inRows, int cols, int outCols) {
  // in [inRows][cols] -> out [cols][outCols], zero-padded past inRows
  size_t i = (size_t)blockIdx.x * blockDim.x + threadIdx.x;
  if (i < (size_t)cols * outCols) {
    int c = (int)(i / outCols), j = (int)(i % outCols);
    out[i] = (j < inRows) ? f2bf(in[(size_t)j * cols + c]) : f2bf(0.f);
  }
}

// ------------------------------- WMMA GEMM ---------------------------------
// C[M,N] = A[M,K](bf16) * Bt[N,K]^T(bf16) + bias.  K staged 64-deep per
// barrier (8 wmma per stage per wave), async global->LDS + prefetch.
// mode 0: f32 row-major (ldc).  mode 1: bf16 head-major [B,NH,S,HD], *scale.

__global__ void __launch_bounds__(128)
gemm_bf16_kernel(const bf16_t* __restrict__ A, const bf16_t* __restrict__ Bt,
                 const float* __restrict__ bias, float* __restrict__ Cf,
                 bf16_t* __restrict__ Cb, int M, int N, int K,
                 int mode, float scale, int ldc) {
  __shared__ __align__(16) bf16_t At[64][64];
  __shared__ __align__(16) bf16_t Bs[64][64];
  const int tid = threadIdx.x, wave = tid >> 5, lane = tid & 31;
  const int lh = lane >> 4, ln = lane & 15;
  const int n0 = blockIdx.x * 64, m0 = blockIdx.y * 64;

  v8f acc[4] = {};
  for (int k0 = 0; k0 < K; k0 += 64) {
    // stage 64x64 A and Bt sub-tiles: 512 16B chunks each, 4 per thread
#pragma unroll
    for (int it = 0; it < 4; ++it) {
      int c = tid + it * 128;
      int row = c >> 3, co = (c & 7) * 8;
      g2l_b128(&A[(size_t)(m0 + row) * K + k0 + co], &At[row][co]);
      g2l_b128(&Bt[(size_t)(n0 + row) * K + k0 + co], &Bs[row][co]);
    }
    if (k0 + 64 < K) {   // prefetch next stage (global_prefetch_b8)
      int row = tid >> 1, co = (tid & 1) * 32;
      __builtin_prefetch(&A[(size_t)(m0 + row) * K + k0 + 64 + co], 0, 1);
      __builtin_prefetch(&Bt[(size_t)(n0 + row) * K + k0 + 64 + co], 0, 1);
    }
    async_wait0();
    __syncthreads();
#pragma unroll
    for (int kk = 0; kk < 2; ++kk) {
      v16bf a = frag_row16(&At[16 * wave + ln][32 * kk], lh);
#pragma unroll
      for (int t = 0; t < 4; ++t) {
        v16bf bb = frag_row16(&Bs[16 * t + ln][32 * kk], lh);
        acc[t] = wmma_bf16(a, bb, acc[t]);
      }
    }
    __syncthreads();
  }
#pragma unroll
  for (int t = 0; t < 4; ++t) {
    int nc = n0 + 16 * t + ln;
    float bvv = bias ? bias[nc] : 0.f;
#pragma unroll
    for (int r = 0; r < 8; ++r) {
      int mr = m0 + 16 * wave + 8 * lh + r;
      float v = acc[t][r] + bvv;
      if (mode == 0) {
        Cf[(size_t)mr * ldc + nc] = v;
      } else {
        int bb_ = mr / S_LEN, ss = mr % S_LEN;
        int hh = nc / HD, dd = nc % HD;
        Cb[(((size_t)bb_ * NH + hh) * S_LEN + ss) * HD + dd] = f2bf(v * scale);
      }
    }
  }
}

// ----------------------------- attention core ------------------------------
// Grid: (S/32, B*NH). Block: 64 threads (2 waves x 16 q-rows). k-tile = 32.

__global__ void __launch_bounds__(64)
attn_relpos_kernel(const bf16_t* __restrict__ Qh, const bf16_t* __restrict__ Kh,
                   const bf16_t* __restrict__ Vh, const float* __restrict__ R,
                   const bf16_t* __restrict__ RVt, bf16_t* __restrict__ Obf) {
  __shared__ __align__(16) bf16_t Ktile[32][HD];   // [k][d]
  __shared__ __align__(16) bf16_t Vt[HD][32];      // [d][k] transposed
  __shared__ __align__(16) bf16_t Pl[2][16][32];   // per-wave P (C->A relayout)
  __shared__ float Wl[32][JPAD];                   // rel-value bucket weights
  __shared__ float alphaS[32];

  const int tid = threadIdx.x;
  const int wave = tid >> 5;
  const int lane = tid & 31;
  const int lh = lane >> 4;
  const int ln = lane & 15;

  const int bh = blockIdx.y;
  const int b = bh / NH, h = bh % NH;
  const int q0 = blockIdx.x * 32;

  for (int i = tid; i < 32 * JPAD; i += 64) ((float*)Wl)[i] = 0.f;

  v16bf qf[2];
  {
    const bf16_t* qrow = Qh + ((size_t)bh * S_LEN + q0 + 16 * wave + ln) * HD;
    qf[0] = frag_row16(qrow, lh);       // d 0..31  (already scaled by 1/8)
    qf[1] = frag_row16(qrow + 32, lh);  // d 32..63
  }

  v8f oacc[4] = {};
  float mrun[8], srun[8];
#pragma unroll
  for (int r = 0; r < 8; ++r) { mrun[r] = -1e30f; srun[r] = 0.f; }

  const float* Rb = R + (size_t)bh * S_LEN * RLD;
  const bf16_t* Kb = Kh + (size_t)bh * S_LEN * HD;
  const bf16_t* Vb = Vh + (size_t)bh * S_LEN * HD;

  __syncthreads();

  for (int k0 = 0; k0 < S_LEN; k0 += 32) {
    // stage K tile (async -> LDS) and transposed V tile
#pragma unroll
    for (int it = 0; it < 4; ++it) {
      int c = tid + it * 64;
      int row = c >> 3;
      int co = (c & 7) * 8;
      g2l_b128(&Kb[(size_t)(k0 + row) * HD + co], &Ktile[row][co]);
      union { uint4 u; bf16_t e[8]; } tv;
      tv.u = *(const uint4*)&Vb[(size_t)(k0 + row) * HD + co];
#pragma unroll
      for (int e = 0; e < 8; ++e) Vt[co + e][row] = tv.e[e];
    }
    if (k0 + 32 < S_LEN) {  // prefetch next K/V tiles
      int row = tid >> 1, co = (tid & 1) * 32;
      __builtin_prefetch(&Kb[(size_t)(k0 + 32 + row) * HD + co], 0, 1);
      __builtin_prefetch(&Vb[(size_t)(k0 + 32 + row) * HD + co], 0, 1);
    }
    async_wait0();
    __syncthreads();

    // scores: two 16-wide column tiles, K-dim = 64 in two WMMA steps
    v8f c0 = {}, c1 = {};
#pragma unroll
    for (int dstep = 0; dstep < 2; ++dstep) {
      v16bf b0 = frag_row16(&Ktile[ln][32 * dstep], lh);
      v16bf b1 = frag_row16(&Ktile[16 + ln][32 * dstep], lh);
      c0 = wmma_bf16(qf[dstep], b0, c0);
      c1 = wmma_bf16(qf[dstep], b1, c1);
    }

    // rel-key bias gather (R is L2-resident; row slice is contiguous-reversed)
#pragma unroll
    for (int r = 0; r < 8; ++r) {
      int qa = q0 + 16 * wave + 8 * lh + r;
      int d0 = qa - (k0 + ln);
      int d1 = qa - (k0 + 16 + ln);
      d0 = d0 < -MAXREL ? -MAXREL : (d0 > MAXREL ? MAXREL : d0);
      d1 = d1 < -MAXREL ? -MAXREL : (d1 > MAXREL ? MAXREL : d1);
      c0[r] += Rb[(size_t)qa * RLD + (d0 + MAXREL)];
      c1[r] += Rb[(size_t)qa * RLD + (d1 + MAXREL)];
    }

    // online softmax: each lane holds 8 rows of one column; reduce over the
    // 16 lanes of its half-group (xor 1,2,4,8 stays inside the group)
#pragma unroll
    for (int r = 0; r < 8; ++r) {
      float m = fmaxf(c0[r], c1[r]);
#pragma unroll
      for (int off = 1; off < 16; off <<= 1) m = fmaxf(m, __shfl_xor(m, off));
      float mnew = fmaxf(mrun[r], m);
      float al = __expf(mrun[r] - mnew);
      c0[r] = __expf(c0[r] - mnew);
      c1[r] = __expf(c1[r] - mnew);
      float ps = c0[r] + c1[r];
#pragma unroll
      for (int off = 1; off < 16; off <<= 1) ps += __shfl_xor(ps, off);
      srun[r] = srun[r] * al + ps;
      mrun[r] = mnew;
      if (ln == 0) alphaS[16 * wave + 8 * lh + r] = al;
      Pl[wave][8 * lh + r][ln] = f2bf(c0[r]);
      Pl[wave][8 * lh + r][16 + ln] = f2bf(c1[r]);
#pragma unroll
      for (int t = 0; t < 4; ++t) oacc[t][r] *= al;   // rescale O accum
    }
    __syncthreads();

    // rescale rel-value buckets by per-row alpha
    for (int i = tid; i < 32 * JPAD; i += 64) {
      int row = i / JPAD;
      ((float*)Wl)[i] *= alphaS[row];
    }
    __syncthreads();

    // scatter-add probabilities into relative-id buckets (ds_add_f32)
#pragma unroll
    for (int r = 0; r < 8; ++r) {
      int qw = 16 * wave + 8 * lh + r;
      int qa = q0 + qw;
      int d0 = qa - (k0 + ln);
      int d1 = qa - (k0 + 16 + ln);
      d0 = d0 < -MAXREL ? -MAXREL : (d0 > MAXREL ? MAXREL : d0);
      d1 = d1 < -MAXREL ? -MAXREL : (d1 > MAXREL ? MAXREL : d1);
      atomicAdd(&Wl[qw][d0 + MAXREL], c0[r]);
      atomicAdd(&Wl[qw][d1 + MAXREL], c1[r]);
    }

    // O += P @ V  (P 16x32 from LDS, V^T rows are B columns)
    {
      v16bf pa = frag_row16(&Pl[wave][ln][0], lh);
#pragma unroll
      for (int t = 0; t < 4; ++t) {
        v16bf bvv = frag_row16(&Vt[16 * t + ln][0], lh);
        oacc[t] = wmma_bf16(pa, bvv, oacc[t]);
      }
    }
    __syncthreads();
  }

  // O += W @ rel_value_table  (K = JPAD; pad columns are zero)
  for (int js = 0; js < JPAD / 32; ++js) {
    v16bf aw;
    const float* wrow = &Wl[16 * wave + ln][32 * js];
#pragma unroll
    for (int e = 0; e < 8; ++e) {
      aw[e] = f2bf(wrow[8 * lh + e]);
      aw[e + 8] = f2bf(wrow[16 + 8 * lh + e]);
    }
#pragma unroll
    for (int t = 0; t < 4; ++t) {
      v16bf bvv = frag_row16(&RVt[(size_t)(16 * t + ln) * JPAD + 32 * js], lh);
      oacc[t] = wmma_bf16(aw, bvv, oacc[t]);
    }
  }

  // normalize by softmax denominator and store concat-head bf16 [B,S,H]
#pragma unroll
  for (int t = 0; t < 4; ++t) {
#pragma unroll
    for (int r = 0; r < 8; ++r) {
      int qa = q0 + 16 * wave + 8 * lh + r;
      int dc = 16 * t + ln;
      float v = oacc[t][r] / srun[r];
      Obf[((size_t)b * S_LEN + qa) * H_DIM + h * HD + dc] = f2bf(v);
    }
  }
}

// ------------------------------- launcher ----------------------------------

extern "C" void kernel_launch(void* const* d_in, const int* in_sizes, int n_in,
                              void* d_out, int out_size, void* d_ws, size_t ws_size,
                              hipStream_t stream) {
  (void)in_sizes; (void)n_in; (void)out_size; (void)ws_size;
  const float* query = (const float*)d_in[0];
  const float* key_  = (const float*)d_in[1];
  const float* value = (const float*)d_in[2];
  const float* Wq = (const float*)d_in[3];
  const float* bq = (const float*)d_in[4];
  const float* Wk = (const float*)d_in[5];
  const float* bk = (const float*)d_in[6];
  const float* Wv = (const float*)d_in[7];
  const float* bv = (const float*)d_in[8];
  const float* Wo = (const float*)d_in[9];
  const float* bo = (const float*)d_in[10];
  const float* rkt = (const float*)d_in[11];  // [257][64]
  const float* rvt = (const float*)d_in[12];  // [257][64]

  // ---- workspace carve-up (~144 MB total) ----
  char* p = (char*)d_ws;
  auto take = [&](size_t bytes) { void* q = (void*)p; p += (bytes + 255) & ~(size_t)255; return q; };
  const size_t nX = (size_t)NTOK * H_DIM;                 // 4,194,304
  bf16_t* Xq  = (bf16_t*)take(nX * 2);
  bf16_t* Xk  = (bf16_t*)take(nX * 2);
  bf16_t* Xv  = (bf16_t*)take(nX * 2);
  bf16_t* WqT = (bf16_t*)take((size_t)H_DIM * H_DIM * 2);
  bf16_t* WkT = (bf16_t*)take((size_t)H_DIM * H_DIM * 2);
  bf16_t* WvT = (bf16_t*)take((size_t)H_DIM * H_DIM * 2);
  bf16_t* WoT = (bf16_t*)take((size_t)H_DIM * H_DIM * 2);
  bf16_t* Qh  = (bf16_t*)take(nX * 2);                    // [B,NH,S,HD]
  bf16_t* Kh  = (bf16_t*)take(nX * 2);
  bf16_t* Vh  = (bf16_t*)take(nX * 2);
  bf16_t* RKTp = (bf16_t*)take((size_t)RLD * HD * 2);     // padded [320][64]
  bf16_t* RVt  = (bf16_t*)take((size_t)HD * JPAD * 2);    // [64][288]
  float*  Rt   = (float*)take((size_t)B_SZ * NH * S_LEN * RLD * 4);  // 80 MB
  bf16_t* Obf  = (bf16_t*)take(nX * 2);

  // 1) conversions
  f32_to_bf16_kernel<<<(unsigned)((nX + 255) / 256), 256, 0, stream>>>(query, Xq, nX);
  f32_to_bf16_kernel<<<(unsigned)((nX + 255) / 256), 256, 0, stream>>>(key_, Xk, nX);
  f32_to_bf16_kernel<<<(unsigned)((nX + 255) / 256), 256, 0, stream>>>(value, Xv, nX);
  const size_t nW = (size_t)H_DIM * H_DIM;
  transpose_f32_to_bf16_kernel<<<(unsigned)((nW + 255) / 256), 256, 0, stream>>>(Wq, WqT, H_DIM, H_DIM);
  transpose_f32_to_bf16_kernel<<<(unsigned)((nW + 255) / 256), 256, 0, stream>>>(Wk, WkT, H_DIM, H_DIM);
  transpose_f32_to_bf16_kernel<<<(unsigned)((nW + 255) / 256), 256, 0, stream>>>(Wv, WvT, H_DIM, H_DIM);
  transpose_f32_to_bf16_kernel<<<(unsigned)((nW + 255) / 256), 256, 0, stream>>>(Wo, WoT, H_DIM, H_DIM);
  pad_rows_bf16_kernel<<<(RLD * HD + 255) / 256, 256, 0, stream>>>(rkt, RKTp, JTOT, HD, RLD);
  transpose_pad_bf16_kernel<<<(HD * JPAD + 255) / 256, 256, 0, stream>>>(rvt, RVt, JTOT, HD, JPAD);

  // 2) projections -> head-major bf16 (Q folded with 1/sqrt(hd) = 0.125)
  dim3 gP(H_DIM / 64, NTOK / 64);
  gemm_bf16_kernel<<<gP, 128, 0, stream>>>(Xq, WqT, bq, nullptr, Qh, NTOK, H_DIM, H_DIM, 1, 0.125f, 0);
  gemm_bf16_kernel<<<gP, 128, 0, stream>>>(Xk, WkT, bk, nullptr, Kh, NTOK, H_DIM, H_DIM, 1, 1.0f, 0);
  gemm_bf16_kernel<<<gP, 128, 0, stream>>>(Xv, WvT, bv, nullptr, Vh, NTOK, H_DIM, H_DIM, 1, 1.0f, 0);

  // 3) rel-key bias table R[bhq, j]
  dim3 gR(RLD / 64, (B_SZ * NH * S_LEN) / 64);
  gemm_bf16_kernel<<<gR, 128, 0, stream>>>(Qh, RKTp, nullptr, Rt, nullptr,
                                           B_SZ * NH * S_LEN, RLD, HD, 0, 1.0f, RLD);

  // 4) attention
  dim3 gA(S_LEN / 32, B_SZ * NH);
  attn_relpos_kernel<<<gA, 64, 0, stream>>>(Qh, Kh, Vh, Rt, RVt, Obf);

  // 5) output projection -> f32 d_out
  gemm_bf16_kernel<<<gP, 128, 0, stream>>>(Obf, WoT, bo, (float*)d_out, nullptr,
                                           NTOK, H_DIM, H_DIM, 0, 1.0f, H_DIM);
}
